// MHA_54812372632202
// MI455X (gfx1250) — compile-verified, matrix operand order checked by reading, not run on hardware
//
#include <hip/hip_runtime.h>

typedef _Float16 f16;
typedef __attribute__((ext_vector_type(16))) _Float16 v16h;
typedef __attribute__((ext_vector_type(8)))  _Float16 v8h;
typedef __attribute__((ext_vector_type(8)))  float    v8f;
typedef __attribute__((ext_vector_type(4)))  unsigned int u32x4;
typedef __attribute__((ext_vector_type(8)))  int i32x8;
typedef __attribute__((ext_vector_type(4)))  int i32x4;

union V16 { v16h v; v8h h[2]; };

static constexpr int S_   = 2048;
static constexpr int E_   = 2048;
static constexpr int H_   = 16;
static constexpr int Dh_  = 128;
static constexpr int NQKV = 3 * E_;   // 6144

#if __has_builtin(__builtin_amdgcn_tensor_load_to_lds)
#define HAVE_TDM 1
#if __has_include(<hip/amd_detail/amd_gfx1250_TDM.h>)
#define TDM_6ARG 1
#endif
#endif

// ---------------------------------------------------------------- convert
__global__ void cvt_f32_f16(const float* __restrict__ in, f16* __restrict__ out, int n) {
  for (int i = blockIdx.x * blockDim.x + threadIdx.x; i < n; i += gridDim.x * blockDim.x)
    out[i] = (f16)in[i];
}

#ifdef HAVE_TDM
// TDM 2D tile load: rows x cols f16 tile from row-major matrix (stride_elems) into LDS
// with 16B padding after each 64B row (LDS row stride = 40 halves).
__device__ __forceinline__ void tdm_load_2d(unsigned lds_addr, const f16* gptr,
                                            int rows, int cols, int stride_elems) {
  const unsigned long long ga = (unsigned long long)(size_t)(const void*)gptr;
  u32x4 g0;
  g0[0] = 1u;                                             // count=1, user descriptor
  g0[1] = lds_addr;                                       // LDS byte address
  g0[2] = (unsigned)(ga & 0xFFFFFFFFu);                   // global addr lo
  g0[3] = (unsigned)((ga >> 32) & 0x01FFFFFFu) | 0x80000000u;  // addr hi | type=2
  i32x8 g1;
  g1[0] = (int)((1u << 16)      // data_size = 2 bytes
              | (1u << 20)      // pad_enable
              | (3u << 22)      // pad_interval: 16 dwords (64B)
              | (3u << 25));    // pad_amount: 4 dwords (16B)
  g1[1] = (int)(((unsigned)cols & 0xFFFFu) << 16);        // tensor_dim0 lo16
  g1[2] = (int)(((unsigned)rows & 0xFFFFu) << 16);        // tensor_dim0 hi=0 | tensor_dim1 lo16
  g1[3] = (int)(((unsigned)cols & 0xFFFFu) << 16);        // tensor_dim1 hi=0 | tile_dim0
  g1[4] = (int)((unsigned)rows & 0xFFFFu);                // tile_dim1 | tile_dim2=0
  g1[5] = stride_elems;                                   // tensor_dim0_stride lo32
  g1[6] = 0;
  g1[7] = 0;
  const i32x4 z4 = {0, 0, 0, 0};
#ifdef TDM_6ARG
  const i32x8 z8 = {0, 0, 0, 0, 0, 0, 0, 0};
  __builtin_amdgcn_tensor_load_to_lds(g0, g1, z4, z4, z8, 0);
#else
  __builtin_amdgcn_tensor_load_to_lds(g0, g1, z4, z4, 0);
#endif
}
#endif

// ---------------------------------------------------------------- GEMM: C[m,n] = sum_k A[m,k]*B[n,k] + bias[n]
// A: M x K, B: N x K (both row-major, K contiguous). Block tile 64(M) x 128(N),
// 256 threads = 8 waves arranged 4(M) x 2(N); each wave computes 16 x 64.
// K-step tiles staged in LDS by the Tensor Data Mover, double-buffered.
template <bool F16OUT>
__global__ __launch_bounds__(256) void gemm_bias_wmma(
    const f16* __restrict__ A, const f16* __restrict__ B,
    const float* __restrict__ bias, void* __restrict__ Cout,
    int K, int ldc) {
  constexpr int LDST = 40;            // padded LDS row stride in halves (64B data + 16B pad)
  constexpr int ASZ  = 64 * LDST;     // 2560 halves
  constexpr int BSZ  = 128 * LDST;    // 5120 halves
  __shared__ alignas(16) f16 smem[2][ASZ + BSZ];

  const int tid  = threadIdx.x;
  const int wave = tid >> 5;
  const int lane = tid & 31;
  const int half = lane >> 4;
  const int l15  = lane & 15;
  const int rowbase = half * 8;
  const int waveM = wave & 3;
  const int waveN = wave >> 2;
  const int m0b = blockIdx.y * 64;
  const int n0b = blockIdx.x * 128;

  v8f acc[4] = {v8f{}, v8f{}, v8f{}, v8f{}};

#ifdef HAVE_TDM
  if (wave == 0) {
    tdm_load_2d((unsigned)(size_t)(void*)&smem[0][0],   A + (size_t)m0b * K,  64, 32, K);
    tdm_load_2d((unsigned)(size_t)(void*)&smem[0][ASZ], B + (size_t)n0b * K, 128, 32, K);
  }
  for (int k0 = 0; k0 < K; k0 += 32) {
    const int p = (k0 >> 5) & 1;
    if (wave == 0) __builtin_amdgcn_s_wait_tensorcnt(0);
    __syncthreads();                       // buffer p valid for all waves
    if (wave == 0 && (k0 + 32) < K) {      // DMA next K-step while computing
      tdm_load_2d((unsigned)(size_t)(void*)&smem[p ^ 1][0],   A + (size_t)m0b * K + k0 + 32,  64, 32, K);
      tdm_load_2d((unsigned)(size_t)(void*)&smem[p ^ 1][ASZ], B + (size_t)n0b * K + k0 + 32, 128, 32, K);
    }
    const f16* sa = &smem[p][0];
    const f16* sb = &smem[p][ASZ];
    V16 a;
    a.h[0] = *(const v8h*)(sa + (waveM * 16 + l15) * LDST + half * 8);
    a.h[1] = *(const v8h*)(sa + (waveM * 16 + l15) * LDST + 16 + half * 8);
#pragma unroll
    for (int t = 0; t < 4; ++t) {
      const f16* brow = sb + (waveN * 64 + t * 16 + l15) * LDST;
      V16 b;
      b.h[0] = *(const v8h*)(brow + half * 8);
      b.h[1] = *(const v8h*)(brow + 16 + half * 8);
      acc[t] = __builtin_amdgcn_wmma_f32_16x16x32_f16(
          false, a.v, false, b.v, (short)0, acc[t], false, false);
    }
  }
#else
  for (int k0 = 0; k0 < K; k0 += 32) {
    __syncthreads();
    {  // cooperative staging: A = 256 16B units, B = 512 16B units
      const int ra = tid >> 2, ca = tid & 3;
      *(v8h*)(&smem[0][ra * LDST + ca * 8]) =
          *(const v8h*)(A + (size_t)(m0b + ra) * K + k0 + ca * 8);
      for (int u = tid; u < 512; u += 256) {
        const int rb = u >> 2, cb = u & 3;
        *(v8h*)(&smem[0][ASZ + rb * LDST + cb * 8]) =
            *(const v8h*)(B + (size_t)(n0b + rb) * K + k0 + cb * 8);
      }
    }
    __syncthreads();
    const f16* sa = &smem[0][0];
    const f16* sb = &smem[0][ASZ];
    V16 a;
    a.h[0] = *(const v8h*)(sa + (waveM * 16 + l15) * LDST + half * 8);
    a.h[1] = *(const v8h*)(sa + (waveM * 16 + l15) * LDST + 16 + half * 8);
#pragma unroll
    for (int t = 0; t < 4; ++t) {
      const f16* brow = sb + (waveN * 64 + t * 16 + l15) * LDST;
      V16 b;
      b.h[0] = *(const v8h*)(brow + half * 8);
      b.h[1] = *(const v8h*)(brow + 16 + half * 8);
      acc[t] = __builtin_amdgcn_wmma_f32_16x16x32_f16(
          false, a.v, false, b.v, (short)0, acc[t], false, false);
    }
  }
#endif

#pragma unroll
  for (int t = 0; t < 4; ++t) {
    const int n = n0b + waveN * 64 + t * 16 + l15;
    const float bv = bias[n];
#pragma unroll
    for (int r = 0; r < 8; ++r) {
      const float val = acc[t][r] + bv;
      const size_t idx = (size_t)(m0b + waveM * 16 + rowbase + r) * ldc + n;
      if (F16OUT) ((f16*)Cout)[idx] = (f16)val;
      else        ((float*)Cout)[idx] = val;
    }
  }
}

// ---------------------------------------------------------------- V transpose: vT[(h*Dh+d)*S + s] = qkv[s*NQKV + 2E + h*Dh + d]
__global__ void transpose_v(const f16* __restrict__ qkv, f16* __restrict__ vT) {
  const int total = H_ * Dh_ * S_;
  for (int i = blockIdx.x * blockDim.x + threadIdx.x; i < total; i += gridDim.x * blockDim.x) {
    const int s  = i & (S_ - 1);
    const int hd = i >> 11;  // i / S_
    vT[i] = qkv[(size_t)s * NQKV + 2 * E_ + hd];
  }
}

// ---------------------------------------------------------------- row reductions across a 16-lane half
__device__ __forceinline__ float rowmax16(float v) {
  v = fmaxf(v, __shfl_xor(v, 1, 32));
  v = fmaxf(v, __shfl_xor(v, 2, 32));
  v = fmaxf(v, __shfl_xor(v, 4, 32));
  v = fmaxf(v, __shfl_xor(v, 8, 32));
  return v;
}
__device__ __forceinline__ float rowsum16(float v) {
  v += __shfl_xor(v, 1, 32);
  v += __shfl_xor(v, 2, 32);
  v += __shfl_xor(v, 4, 32);
  v += __shfl_xor(v, 8, 32);
  return v;
}

// ---------------------------------------------------------------- flash attention: one wave = (head, 16-query tile)
__global__ __launch_bounds__(128) void flash_attn(
    const f16* __restrict__ qkv, const f16* __restrict__ vT, f16* __restrict__ ao) {
  __shared__ f16 pbuf[4][16 * 32];

  const int wave = threadIdx.x >> 5;
  const int lane = threadIdx.x & 31;
  const int half = lane >> 4;
  const int l15  = lane & 15;
  const int rowbase = half * 8;

  const int widx  = blockIdx.x * 4 + wave;
  const int head  = widx >> 7;     // / (S_/16)
  const int qbase = (widx & 127) * 16;
  const float scale = 0.088388347648318447f;  // 1/sqrt(128)

  // Q fragments (A operand), K=128 in 4 chunks of 32
  const f16* qp = qkv + (size_t)(qbase + l15) * NQKV + head * Dh_;
  V16 qa[4];
#pragma unroll
  for (int c = 0; c < 4; ++c) {
    qa[c].h[0] = *(const v8h*)(qp + c * 32 + half * 8);
    qa[c].h[1] = *(const v8h*)(qp + c * 32 + 16 + half * 8);
  }

  v8f o[8] = {v8f{}, v8f{}, v8f{}, v8f{}, v8f{}, v8f{}, v8f{}, v8f{}};
  float mi[8], li[8];
#pragma unroll
  for (int r = 0; r < 8; ++r) { mi[r] = -1e30f; li[r] = 0.0f; }

  const int ntiles = (qbase + 16 + 31) >> 5;  // causal: keys 0 .. qbase+15
  f16* pb = pbuf[wave];

  for (int kt = 0; kt < ntiles; ++kt) {
    const int k0 = kt * 32;
    // ---- scores: two 16-key subtiles, K=128 reduction
    v8f sfrag[2];
#pragma unroll
    for (int sub = 0; sub < 2; ++sub) {
      v8f accs = {};
      const f16* kp = qkv + (size_t)(k0 + sub * 16 + l15) * NQKV + E_ + head * Dh_;
#pragma unroll
      for (int c = 0; c < 4; ++c) {
        V16 kb;
        kb.h[0] = *(const v8h*)(kp + c * 32 + half * 8);
        kb.h[1] = *(const v8h*)(kp + c * 32 + 16 + half * 8);
        accs = __builtin_amdgcn_wmma_f32_16x16x32_f16(
            false, qa[c].v, false, kb.v, (short)0, accs, false, false);
      }
      sfrag[sub] = accs;
    }
    // ---- scale + causal mask + online softmax stats
    float corr[8];
#pragma unroll
    for (int r = 0; r < 8; ++r) {
      const int row = qbase + rowbase + r;
      float s0 = sfrag[0][r] * scale + ((k0 + l15)      > row ? -10000.0f : 0.0f);
      float s1 = sfrag[1][r] * scale + ((k0 + 16 + l15) > row ? -10000.0f : 0.0f);
      const float nm = fmaxf(mi[r], rowmax16(fmaxf(s0, s1)));
      corr[r] = __expf(mi[r] - nm);
      mi[r] = nm;
      const float p0 = __expf(s0 - nm);
      const float p1 = __expf(s1 - nm);
      sfrag[0][r] = p0;
      sfrag[1][r] = p1;
      li[r] = li[r] * corr[r] + rowsum16(p0 + p1);
    }
#pragma unroll
    for (int j = 0; j < 8; ++j)
#pragma unroll
      for (int r = 0; r < 8; ++r) o[j][r] *= corr[r];

    // ---- restage P through LDS: C-layout (lane=col, vgpr=row) -> A-layout
#pragma unroll
    for (int r = 0; r < 8; ++r) {
      pb[(rowbase + r) * 32 + l15]      = (f16)sfrag[0][r];
      pb[(rowbase + r) * 32 + 16 + l15] = (f16)sfrag[1][r];
    }
    asm volatile("s_wait_dscnt 0" ::: "memory");
    V16 pa;
    pa.h[0] = *(const v8h*)(pb + l15 * 32 + half * 8);
    pa.h[1] = *(const v8h*)(pb + l15 * 32 + 16 + half * 8);

    // ---- O += P(16x32) * V(32x128): 8 N-chunks, B from transposed V
#pragma unroll
    for (int j = 0; j < 8; ++j) {
      const f16* vp = vT + ((size_t)head * Dh_ + j * 16 + l15) * S_ + k0;
      V16 vb;
      vb.h[0] = *(const v8h*)(vp + half * 8);
      vb.h[1] = *(const v8h*)(vp + 16 + half * 8);
      o[j] = __builtin_amdgcn_wmma_f32_16x16x32_f16(
          false, pa.v, false, vb.v, (short)0, o[j], false, false);
    }
  }

  // ---- epilogue: normalize and store f16
#pragma unroll
  for (int j = 0; j < 8; ++j) {
#pragma unroll
    for (int r = 0; r < 8; ++r) {
      const float val = o[j][r] / li[r];
      ao[(size_t)(qbase + rowbase + r) * E_ + head * Dh_ + j * 16 + l15] = (f16)val;
    }
  }
}

// ---------------------------------------------------------------- launcher
extern "C" void kernel_launch(void* const* d_in, const int* in_sizes, int n_in,
                              void* d_out, int out_size, void* d_ws, size_t ws_size,
                              hipStream_t stream) {
  const float* x      = (const float*)d_in[0];
  const float* w_qkv  = (const float*)d_in[1];
  const float* b_qkv  = (const float*)d_in[2];
  const float* w_out  = (const float*)d_in[3];
  const float* b_out  = (const float*)d_in[4];

  char* ws = (char*)d_ws;
  f16* x16    = (f16*)(ws + 0);
  f16* wqkv16 = (f16*)(ws + 8388608);    //  3E*E f16 = 24MB
  f16* wout16 = (f16*)(ws + 33554432);   //  E*E  f16 =  8MB
  f16* qkv16  = (f16*)(ws + 41943040);   //  S*3E f16 = 24MB
  f16* vT16   = (f16*)(ws + 67108864);   //  H*Dh*S   =  8MB
  f16* ao16   = (f16*)(ws + 75497472);   //  S*E      =  8MB

  // 1. fp32 -> f16 conversions
  cvt_f32_f16<<<2048, 256, 0, stream>>>(x,     x16,    S_ * E_);
  cvt_f32_f16<<<2048, 256, 0, stream>>>(w_qkv, wqkv16, NQKV * E_);
  cvt_f32_f16<<<2048, 256, 0, stream>>>(w_out, wout16, E_ * E_);

  // 2. qkv = x @ w_qkv^T + b_qkv  (S x 6144, f16)
  gemm_bias_wmma<true><<<dim3(NQKV / 128, S_ / 64), 256, 0, stream>>>(
      x16, wqkv16, b_qkv, qkv16, E_, NQKV);

  // 3. transpose V per head -> (H, Dh, S)
  transpose_v<<<2048, 256, 0, stream>>>(qkv16, vT16);

  // 4. causal flash attention -> ao16 (S x E, f16)
  flash_attn<<<(H_ * (S_ / 16)) / 4, 128, 0, stream>>>(qkv16, vT16, ao16);

  // 5. out = ao @ w_out^T + b_out  (S x E, fp32)
  gemm_bias_wmma<false><<<dim3(E_ / 128, S_ / 64), 256, 0, stream>>>(
      ao16, wout16, b_out, d_out, E_, E_);
}
